// PointConvSA_1666447310848
// MI455X (gfx1250) — matrix operand order repaired and checked by reading, not compiled, not run on hardware
//
#include <hip/hip_runtime.h>
#include <hip/hip_bf16.h>
#include <math.h>

// ---------------------------------------------------------------------------
// PointConvSA for MI455X (gfx1250, wave32, WMMA)
// Inputs (setup_inputs order):
//  0 xyz      (4,4096,3)  f32
//  1 feature  (4,4096,16) f32
//  2 w_mlp0   (19,128)    f32
//  3 w_mlp1   (128,128)   f32
//  4 w_mlp2   (128,256)   f32   (only column 0 is ever used downstream!)
//  5 w_wn     (3,32)      f32
//  6 w_nl0    (1,16)      f32
//  7 w_nl1    (16,1)      f32
//  8 w_np     (32,256)    f32
// Output: concat( new_xyz (4,1024,3), out (4,1024,256) ) f32
// ---------------------------------------------------------------------------

typedef __bf16 bf16;
typedef __attribute__((ext_vector_type(16))) __bf16 v16bf;
typedef __attribute__((ext_vector_type(8)))  __bf16 v8bf;
typedef __attribute__((ext_vector_type(8)))  float  v8f;

#define B_    4
#define N_    4096
#define C_    16
#define NP_   1024
#define K_    32
#define KDEK_ 128
// radius^2 = 0.01 ; sigma = 0.05 ; -0.5/sigma = -10 ;
// logC = -1.5*(log(sigma)+log(2*pi)) = 1.7367766
#define R2_     0.01f
#define NEGHALFINVSIG_ (-10.0f)
#define LOGC_   1.7367766f

__device__ __forceinline__ v16bf cat8(v8bf lo, v8bf hi) {
  return __builtin_shufflevector(lo, hi, 0,1,2,3,4,5,6,7,8,9,10,11,12,13,14,15);
}
__device__ __forceinline__ v8f zero8() {
  v8f z = {0.f,0.f,0.f,0.f,0.f,0.f,0.f,0.f};
  return z;
}

// ---------------------------------------------------------------------------
// Weight prep: convert W0/W1 to bf16 packed in WMMA B-fragment lane order.
// B-fragment as packed here (16x16x32 bf16, contiguous K order, A uses the same): lane holds column N = lane&15,
//   lanes 0-15 -> K = 0..15, lanes 16-31 -> K = 16..31 of the 32-K tile,
//   16 contiguous bf16 per lane per fragment.
// Wt0: 8 N-tiles (K-tile 0 only, rows >=19 zero-padded). entry = nt*32+lane.
// Wt1: 4 K-tiles x 8 N-tiles. entry = (kt*8+nt)*32+lane.
// Also extract w_mlp2[:,0] (128 f32).
// ---------------------------------------------------------------------------
__global__ void prep_pack(const float* __restrict__ w0,
                          const float* __restrict__ w1,
                          const float* __restrict__ w2,
                          bf16* __restrict__ Wt0,
                          bf16* __restrict__ Wt1,
                          float* __restrict__ w2c0) {
  const int tid = threadIdx.x;          // 256 threads
  {
    const int e = tid;                  // Wt0 entry
    const int lane = e & 31;
    const int nt   = e >> 5;            // 0..7
    const int n    = nt * 16 + (lane & 15);
    const int kb   = (lane >> 4) * 16;
#pragma unroll
    for (int j = 0; j < 16; ++j) {
      const int k = kb + j;
      const float v = (k < 19) ? w0[k * 128 + n] : 0.0f;
      Wt0[e * 16 + j] = (bf16)v;
    }
  }
#pragma unroll
  for (int q = 0; q < 4; ++q) {
    const int e    = tid + q * 256;     // 0..1023
    const int lane = e & 31;
    const int frag = e >> 5;            // kt*8+nt
    const int nt   = frag & 7;
    const int kt   = frag >> 3;
    const int n    = nt * 16 + (lane & 15);
    const int kb   = kt * 32 + (lane >> 4) * 16;
#pragma unroll
    for (int j = 0; j < 16; ++j)
      Wt1[e * 16 + j] = (bf16)w1[(kb + j) * 128 + n];
  }
  if (tid < 128) w2c0[tid] = w2[tid * 256 + 0];
}

// ---------------------------------------------------------------------------
// Farthest point sampling: one block per batch, serial 1023 steps with a
// block-wide argmax (tie -> lowest index, matching jnp.argmax).
// Writes new_xyz directly into d_out's first region.
// ---------------------------------------------------------------------------
__global__ __launch_bounds__(256)
void fps_kernel(const float* __restrict__ xyz, float* __restrict__ newxyz) {
  const int b   = blockIdx.x;
  const int tid = threadIdx.x;          // 256
  const float* X = xyz + (size_t)b * N_ * 3;

  __shared__ float mind[N_];
  __shared__ float rv[256];
  __shared__ int   ri[256];
  __shared__ float pcur[3];

  for (int n = tid; n < N_; n += 256) mind[n] = 3.4e38f;
  if (tid == 0) {
    pcur[0] = X[0]; pcur[1] = X[1]; pcur[2] = X[2];
    newxyz[((size_t)b * NP_ + 0) * 3 + 0] = X[0];
    newxyz[((size_t)b * NP_ + 0) * 3 + 1] = X[1];
    newxyz[((size_t)b * NP_ + 0) * 3 + 2] = X[2];
  }
  __syncthreads();

  for (int i = 1; i < NP_; ++i) {
    const float px = pcur[0], py = pcur[1], pz = pcur[2];
    float bv = -1.0f; int bi = 0;
    for (int n = tid; n < N_; n += 256) {
      const float dx = X[n*3+0] - px;
      const float dy = X[n*3+1] - py;
      const float dz = X[n*3+2] - pz;
      const float d  = dx*dx + dy*dy + dz*dz;
      float m = mind[n];
      m = fminf(m, d);
      mind[n] = m;
      if (m > bv || (m == bv && n < bi)) { bv = m; bi = n; }
    }
    rv[tid] = bv; ri[tid] = bi;
    __syncthreads();
#pragma unroll
    for (int s = 128; s > 0; s >>= 1) {
      if (tid < s) {
        const float v2 = rv[tid + s]; const int i2 = ri[tid + s];
        if (v2 > rv[tid] || (v2 == rv[tid] && i2 < ri[tid])) {
          rv[tid] = v2; ri[tid] = i2;
        }
      }
      __syncthreads();
    }
    if (tid == 0) {
      const int nx = ri[0];
      const float x0 = X[nx*3+0], y0 = X[nx*3+1], z0 = X[nx*3+2];
      pcur[0] = x0; pcur[1] = y0; pcur[2] = z0;
      newxyz[((size_t)b * NP_ + i) * 3 + 0] = x0;
      newxyz[((size_t)b * NP_ + i) * 3 + 1] = y0;
      newxyz[((size_t)b * NP_ + i) * 3 + 2] = z0;
    }
    __syncthreads();
  }
}

// ---------------------------------------------------------------------------
// KDE ball density -> inv_density[b][n]. One thread per point; neighbors
// tiled through LDS. In-ball neighbors taken in ascending index order up to
// KDE_K (matches reference's keyed top_k + count-correction exactly).
// ---------------------------------------------------------------------------
__global__ __launch_bounds__(256)
void kde_kernel(const float* __restrict__ xyz, float* __restrict__ invden) {
  const int gidx = blockIdx.x * 256 + threadIdx.x;  // 0..B*N-1, block within one b
  const int b = gidx >> 12;
  const int i = gidx & (N_ - 1);
  const float* X = xyz + (size_t)b * N_ * 3;
  const float qx = X[i*3+0], qy = X[i*3+1], qz = X[i*3+2];

  __shared__ float tx[256], ty[256], tz[256];

  int   cntAll = 0, taken = 0;
  float total  = 0.0f;
  for (int t = 0; t < N_ / 256; ++t) {
    const int j0 = t * 256 + threadIdx.x;
    tx[threadIdx.x] = X[j0*3+0];
    ty[threadIdx.x] = X[j0*3+1];
    tz[threadIdx.x] = X[j0*3+2];
    __syncthreads();
    for (int j = 0; j < 256; ++j) {
      const float dx = tx[j] - qx, dy = ty[j] - qy, dz = tz[j] - qz;
      const float d2 = dx*dx + dy*dy + dz*dz;
      if (d2 < R2_) {
        ++cntAll;
        if (taken < KDEK_) {
          ++taken;
          total += expf(NEGHALFINVSIG_ * d2 + LOGC_);
        }
      }
    }
    __syncthreads();
  }
  const int cnt = (cntAll < KDEK_) ? cntAll : KDEK_;
  invden[gidx] = (float)cnt / total;
}

// ---------------------------------------------------------------------------
// kNN(32): one thread per query, insertion list (stable: earlier index wins
// on ties, like top_k).
// ---------------------------------------------------------------------------
__global__ __launch_bounds__(64)
void knn_kernel(const float* __restrict__ xyz,
                const float* __restrict__ newxyz,
                int* __restrict__ knn) {
  const int g = blockIdx.x * 64 + threadIdx.x;      // 0..B*NP-1
  const int b = g >> 10;
  const float* X = xyz + (size_t)b * N_ * 3;
  const float qx = newxyz[g*3+0], qy = newxyz[g*3+1], qz = newxyz[g*3+2];

  float bd[K_]; int bi[K_];
#pragma unroll
  for (int k = 0; k < K_; ++k) { bd[k] = 3.4e38f; bi[k] = 0; }

  for (int j = 0; j < N_; ++j) {
    const float dx = X[j*3+0] - qx;
    const float dy = X[j*3+1] - qy;
    const float dz = X[j*3+2] - qz;
    const float d2 = dx*dx + dy*dy + dz*dz;
    if (d2 < bd[K_-1]) {
      int pos = K_ - 1;
      while (pos > 0 && bd[pos-1] > d2) {
        bd[pos] = bd[pos-1]; bi[pos] = bi[pos-1]; --pos;
      }
      bd[pos] = d2; bi[pos] = j;
    }
  }
#pragma unroll
  for (int k = 0; k < K_; ++k) knn[(size_t)g * K_ + k] = bi[k];
}

// ---------------------------------------------------------------------------
// Main fused group kernel: 1 wave per (b,p) group, 4 waves per block.
//  - gather 32x19 group features -> LDS bf16 (padded to 32)
//  - layer0: WMMA 32x128 = X(32x32) @ W0  -> relu -> bf16 LDS
//  - layer1: WMMA 32x128 accumulate over 4 K-tiles; fused fold of
//            relu(.)*w_mlp2[:,0] into h0 (only channel 0 survives!)
//  - density-scale scalar MLP, weight-net, k-contraction, final 32->256.
// ---------------------------------------------------------------------------
__global__ __launch_bounds__(128)
void pointconv_kernel(const float* __restrict__ xyz,
                      const float* __restrict__ feat,
                      const float* __restrict__ newxyz,
                      const float* __restrict__ invden,
                      const int*   __restrict__ knn,
                      const bf16*  __restrict__ Wt0,
                      const bf16*  __restrict__ Wt1,
                      const float* __restrict__ w2c0,
                      const float* __restrict__ wwn,
                      const float* __restrict__ wnl0,
                      const float* __restrict__ wnl1,
                      const float* __restrict__ wnp,
                      float* __restrict__ out) {
  const int lane = threadIdx.x & 31;
  const int wv   = threadIdx.x >> 5;          // 0..3
  const int g    = blockIdx.x * 4 + wv;       // 0..B*NP-1
  const int b    = g >> 10;

  __shared__ bf16  Xl[4][32 * 32];            // padded group features
  __shared__ bf16  H1[4][32 * 128];           // layer0 activations (bf16)
  __shared__ float gx3[4][32][3];
  __shared__ float sArr[4][32];
  __shared__ float pts0[4][32];
  __shared__ float ov[4][32];

  // ---- Phase 1: gather + density scalar chain (lane == k) ----
  {
    const int k   = lane;
    const int idx = knn[(size_t)g * K_ + k];
    const float qx = newxyz[g*3+0], qy = newxyz[g*3+1], qz = newxyz[g*3+2];
    const size_t pt = (size_t)b * N_ + idx;
    const float gx = xyz[pt*3+0] - qx;
    const float gy = xyz[pt*3+1] - qy;
    const float gz = xyz[pt*3+2] - qz;
    gx3[wv][k][0] = gx; gx3[wv][k][1] = gy; gx3[wv][k][2] = gz;
    bf16* xr = &Xl[wv][k * 32];
    xr[0] = (bf16)gx; xr[1] = (bf16)gy; xr[2] = (bf16)gz;
#pragma unroll
    for (int c = 0; c < C_; ++c) xr[3 + c] = (bf16)feat[pt * C_ + c];
#pragma unroll
    for (int c = 19; c < 32; ++c) xr[c] = (bf16)0.0f;

    const float gd = invden[pt];
    float mx = gd;
#pragma unroll
    for (int off = 16; off > 0; off >>= 1)
      mx = fmaxf(mx, __shfl_xor(mx, off, 32));
    const float d0 = gd / mx;
    float s = 0.0f;
#pragma unroll
    for (int j = 0; j < 16; ++j)
      s += fmaxf(d0 * wnl0[j], 0.0f) * wnl1[j];
    sArr[wv][k] = fmaxf(s, 0.0f);
  }
  __syncthreads();

  const int half = lane >> 4;   // which 16-lane half
  const int col  = lane & 15;   // N within tile / row selector

  // ---- Phase 2: layer0  (X[32x32] @ W0[32x128]) ----
#pragma unroll
  for (int mt = 0; mt < 2; ++mt) {
    const int row = mt * 16 + col;
    const int kb  = half * 16;
    // A fragment contiguous K order (matches B packing): elems 0..15 = K kb..kb+15
    const v8bf lo = *(const v8bf*)&Xl[wv][row * 32 + kb];
    const v8bf hi = *(const v8bf*)&Xl[wv][row * 32 + kb + 8];
    const v16bf a = cat8(lo, hi);
#pragma unroll
    for (int nt = 0; nt < 8; ++nt) {
      const v16bf bf = *(const v16bf*)&Wt0[(nt * 32 + lane) * 16];
      v8f d = zero8();
      d = __builtin_amdgcn_wmma_f32_16x16x32_bf16(false, a, false, bf,
                                                  (short)0, d, false, false);
#pragma unroll
      for (int r = 0; r < 8; ++r) {
        const float v = fmaxf(d[r], 0.0f);
        H1[wv][(mt * 16 + half * 8 + r) * 128 + nt * 16 + col] = (bf16)v;
      }
    }
  }
  __syncthreads();

  // ---- Phase 3: layer1 (H1 @ W1) + fused layer2 column-0 reduction ----
  float w2l[8];
#pragma unroll
  for (int nt = 0; nt < 8; ++nt) w2l[nt] = w2c0[nt * 16 + col];

#pragma unroll
  for (int mt = 0; mt < 2; ++mt) {
    const int row = mt * 16 + col;
    v16bf afr[4];
#pragma unroll
    for (int kt = 0; kt < 4; ++kt) {
      const int kb2 = kt * 32 + half * 16;
      const v8bf lo = *(const v8bf*)&H1[wv][row * 128 + kb2];
      const v8bf hi = *(const v8bf*)&H1[wv][row * 128 + kb2 + 8];
      afr[kt] = cat8(lo, hi);
    }
    float p8[8];
#pragma unroll
    for (int r = 0; r < 8; ++r) p8[r] = 0.0f;

#pragma unroll
    for (int nt = 0; nt < 8; ++nt) {
      v8f d = zero8();
#pragma unroll
      for (int kt = 0; kt < 4; ++kt) {
        const v16bf bf = *(const v16bf*)&Wt1[((kt * 8 + nt) * 32 + lane) * 16];
        d = __builtin_amdgcn_wmma_f32_16x16x32_bf16(false, afr[kt], false, bf,
                                                    (short)0, d, false, false);
      }
      const float w2v = w2l[nt];
#pragma unroll
      for (int r = 0; r < 8; ++r) p8[r] += fmaxf(d[r], 0.0f) * w2v;
    }
    // butterfly-reduce across each 16-lane half (N dimension)
#pragma unroll
    for (int off = 1; off < 16; off <<= 1)
#pragma unroll
      for (int r = 0; r < 8; ++r) p8[r] += __shfl_xor(p8[r], off, 32);
    if (col == 0) {
#pragma unroll
      for (int r = 0; r < 8; ++r) {
        const int kk = mt * 16 + half * 8 + r;
        const float h0 = fmaxf(p8[r], 0.0f);
        pts0[wv][kk] = h0 * sArr[wv][kk];
      }
    }
  }
  __syncthreads();

  // ---- Phase 4: weight-net + k-contraction (lane == w) ----
  {
    const int w = lane;
    const float a0 = wwn[0 * 32 + w];
    const float a1 = wwn[1 * 32 + w];
    const float a2 = wwn[2 * 32 + w];
    float acc = 0.0f;
#pragma unroll
    for (int kk = 0; kk < K_; ++kk) {
      const float wn = fmaxf(gx3[wv][kk][0] * a0 + gx3[wv][kk][1] * a1 +
                             gx3[wv][kk][2] * a2, 0.0f);
      acc += pts0[wv][kk] * wn;
    }
    ov[wv][w] = acc;
  }
  __syncthreads();

  // ---- Phase 5: final projection 32 -> 256 (coalesced f = lane + 32q) ----
  {
    float accf[8];
#pragma unroll
    for (int q = 0; q < 8; ++q) accf[q] = 0.0f;
    for (int w = 0; w < 32; ++w) {
      const float o = ov[wv][w];
      const float* wr = &wnp[w * 256];
#pragma unroll
      for (int q = 0; q < 8; ++q) accf[q] += o * wr[lane + q * 32];
    }
    float* od = out + (size_t)g * 256;
#pragma unroll
    for (int q = 0; q < 8; ++q) od[lane + q * 32] = fmaxf(accf[q], 0.0f);
  }
}

// ---------------------------------------------------------------------------
extern "C" void kernel_launch(void* const* d_in, const int* in_sizes, int n_in,
                              void* d_out, int out_size, void* d_ws, size_t ws_size,
                              hipStream_t stream) {
  (void)in_sizes; (void)n_in; (void)out_size; (void)ws_size;
  const float* xyz  = (const float*)d_in[0];
  const float* feat = (const float*)d_in[1];
  const float* w0   = (const float*)d_in[2];
  const float* w1   = (const float*)d_in[3];
  const float* w2   = (const float*)d_in[4];
  const float* wwn  = (const float*)d_in[5];
  const float* wnl0 = (const float*)d_in[6];
  const float* wnl1 = (const float*)d_in[7];
  const float* wnp  = (const float*)d_in[8];

  char* ws = (char*)d_ws;
  float* invden = (float*)(ws);                                    //  64 KB
  int*   knn    = (int*)  (ws + 65536);                            // 512 KB
  bf16*  Wt0    = (bf16*) (ws + 65536 + 524288);                   //   8 KB
  bf16*  Wt1    = (bf16*) (ws + 65536 + 524288 + 8192);            //  32 KB
  float* w2c0   = (float*)(ws + 65536 + 524288 + 8192 + 32768);    // 512 B

  float* newxyz = (float*)d_out;                 // (B,NP,3)
  float* outp   = (float*)d_out + B_ * NP_ * 3;  // (B,NP,256)

  prep_pack<<<1, 256, 0, stream>>>(w0, w1, w2, Wt0, Wt1, w2c0);
  fps_kernel<<<B_, 256, 0, stream>>>(xyz, newxyz);
  kde_kernel<<<(B_ * N_) / 256, 256, 0, stream>>>(xyz, invden);
  knn_kernel<<<(B_ * NP_) / 64, 64, 0, stream>>>(xyz, newxyz, knn);
  pointconv_kernel<<<(B_ * NP_) / 4, 128, 0, stream>>>(
      xyz, feat, newxyz, invden, knn, Wt0, Wt1, w2c0,
      wwn, wnl0, wnl1, wnp, outp);
}
